// Attention_82317343195240
// MI455X (gfx1250) — matrix operand order repaired and checked by reading, not compiled, not run on hardware
//
#include <hip/hip_runtime.h>
#include <stdint.h>

typedef __attribute__((ext_vector_type(16))) __bf16 v16bf;
typedef __attribute__((ext_vector_type(8)))  float  v8f;

// Fragment viewed as two 16-byte chunks (CDNA5 ISA 7.12.2 A-layout runs
// k=8*half+0..7 and k=16+8*half+0..7 are contiguous; B-layout uses the same runs).
union Frag16 {
    v16bf  v;
    float4 f4[2];
};

// ---------------------------------------------------------------------------
// CDNA5 async global->LDS copy (ASYNCcnt-tracked), ISA §15.18.3 op 98.
// LDS generic pointers keep the byte offset in the low 32 bits.
// ---------------------------------------------------------------------------
__device__ __forceinline__ void async_copy_b128(void* lds_ptr, const void* gptr) {
    uint32_t lds_off = (uint32_t)(uintptr_t)lds_ptr;
    asm volatile("global_load_async_to_lds_b128 %0, %1, off"
                 :: "v"(lds_off), "v"((uint64_t)(uintptr_t)gptr)
                 : "memory");
}
__device__ __forceinline__ void wait_async_0() {
    asm volatile("s_wait_asynccnt 0" ::: "memory");
}
__device__ __forceinline__ void wait_async_4() {
    asm volatile("s_wait_asynccnt 4" ::: "memory");
}
__device__ __forceinline__ void wait_async_8() {
    asm volatile("s_wait_asynccnt 8" ::: "memory");
}

// f32 -> bf16 elementwise conversion
__global__ __launch_bounds__(256) void f32_to_bf16_kernel(const float* __restrict__ in,
                                                          __bf16* __restrict__ out, int n) {
    int i = blockIdx.x * 256 + threadIdx.x;
    if (i < n) out[i] = (__bf16)in[i];
}

// Tiled transpose + convert: in f32 [K][N] -> out bf16 [N][K]
__global__ __launch_bounds__(256) void transpose_cvt_kernel(const float* __restrict__ in,
                                                            __bf16* __restrict__ out,
                                                            int K, int N) {
    __shared__ float tile[32][33];
    const int kb = blockIdx.y * 32, nb = blockIdx.x * 32;
    const int tx = threadIdx.x & 31, ty = threadIdx.x >> 5;   // 32 x 8
#pragma unroll
    for (int i = 0; i < 32; i += 8)
        tile[ty + i][tx] = in[(size_t)(kb + ty + i) * N + nb + tx];
    __syncthreads();
#pragma unroll
    for (int i = 0; i < 32; i += 8)
        out[(size_t)(nb + ty + i) * K + kb + tx] = (__bf16)tile[tx][ty + i];
}

// ---------------------------------------------------------------------------
// bf16 GEMM: C[M,N] = A[M,K] @ B[K,N] with B given PRE-TRANSPOSED as Bt[N][K].
// Block tile 128x128, K step 32, 8 waves each computing 32x64 (8 WMMA/step).
// Double-buffered LDS filled by async global->LDS b128 copies.
// OUT_MODE: 0 = bf16 C[m][n], 1 = f32 C[m][n], 2 = bf16 transposed C[n][m].
// ---------------------------------------------------------------------------
template <int OUT_MODE>
__global__ __launch_bounds__(256) void gemm_bf16_kernel(const __bf16* __restrict__ A,
                                                        const __bf16* __restrict__ Bt,
                                                        void* __restrict__ Cout,
                                                        int M, int N, int K) {
    __shared__ __align__(16) __bf16 As[2][128][40];   // [m][k] (32 k + 8 pad)
    __shared__ __align__(16) __bf16 Bs[2][128][40];   // [n][k]

    const int tid  = threadIdx.x;
    const int lane = tid & 31;
    const int wave = tid >> 5;
    const int half = lane >> 4;
    const int mr   = lane & 15;
    const int nc   = lane & 15;
    const int wm   = wave >> 1;
    const int wn   = wave & 1;
    const int mBase = blockIdx.y * 128;
    const int nBase = blockIdx.x * 128;

    auto issue = [&](int k0, int buf) {
#pragma unroll
        for (int i = 0; i < 2; ++i) {
            int e   = (tid + i * 256) * 8;
            int row = e >> 5;              // 0..127
            int col = e & 31;              // 0,8,16,24
            async_copy_b128(&As[buf][row][col], &A [(size_t)(mBase + row) * K + k0 + col]);
            async_copy_b128(&Bs[buf][row][col], &Bt[(size_t)(nBase + row) * K + k0 + col]);
        }
    };

    v8f acc[2][4] = {};
    const int nk = K / 32;
    issue(0, 0);
    for (int kt = 0; kt < nk; ++kt) {
        const int buf = kt & 1;
        if (kt + 1 < nk) { issue((kt + 1) * 32, buf ^ 1); wait_async_4(); }
        else             { wait_async_0(); }
        __syncthreads();

        Frag16 af[2], bfr[4];
#pragma unroll
        for (int mi = 0; mi < 2; ++mi) {
            const __bf16* rp = &As[buf][wm * 32 + mi * 16 + mr][0];
            af[mi].f4[0] = *(const float4*)&rp[8 * half];
            af[mi].f4[1] = *(const float4*)&rp[16 + 8 * half];
        }
#pragma unroll
        for (int t = 0; t < 4; ++t) {
            const __bf16* rp = &Bs[buf][wn * 64 + t * 16 + nc][0];
            bfr[t].f4[0] = *(const float4*)&rp[8 * half];
            bfr[t].f4[1] = *(const float4*)&rp[16 + 8 * half];
        }
#pragma unroll
        for (int mi = 0; mi < 2; ++mi)
#pragma unroll
            for (int t = 0; t < 4; ++t)
                acc[mi][t] = __builtin_amdgcn_wmma_f32_16x16x32_bf16(
                    false, af[mi].v, false, bfr[t].v, (short)0, acc[mi][t], false, false);
        __syncthreads();
    }

#pragma unroll
    for (int mi = 0; mi < 2; ++mi)
#pragma unroll
        for (int t = 0; t < 4; ++t)
#pragma unroll
            for (int r = 0; r < 8; ++r) {
                int m = mBase + wm * 32 + mi * 16 + r + 8 * half;
                int n = nBase + wn * 64 + t * 16 + nc;
                if (OUT_MODE == 1)
                    ((float*)Cout)[(size_t)m * N + n] = acc[mi][t][r];
                else if (OUT_MODE == 2)
                    ((__bf16*)Cout)[(size_t)n * M + m] = (__bf16)acc[mi][t][r];
                else
                    ((__bf16*)Cout)[(size_t)m * N + n] = (__bf16)acc[mi][t][r];
            }
}

// ---------------------------------------------------------------------------
// RoPE (interleaved pairs) with fused output scaling, in-place on bf16
// [B*S, nH*128]. For q: scale = log2(e)/sqrt(HD) (exp2-domain softmax).
// ---------------------------------------------------------------------------
__global__ __launch_bounds__(256) void rope_kernel(__bf16* __restrict__ t,
                                                   const float* __restrict__ cs,
                                                   const float* __restrict__ sn,
                                                   int nH, int total, float scale) {
    int i = blockIdx.x * 256 + threadIdx.x;
    if (i >= total) return;
    const int HD2 = 64, S = 2048;
    int d2  = i & 63;
    int h   = (i >> 6) % nH;
    int row = i / (64 * nH);       // b*S + s
    int s   = row & (S - 1);
    float c  = cs[s * HD2 + d2];
    float sv = sn[s * HD2 + d2];
    size_t idx = (size_t)row * (nH * 128) + h * 128 + 2 * d2;
    float t0 = (float)t[idx], t1 = (float)t[idx + 1];
    t[idx]     = (__bf16)((t0 * c - t1 * sv) * scale);
    t[idx + 1] = (__bf16)((t0 * sv + t1 * c) * scale);
}

// ---------------------------------------------------------------------------
// Flash attention (causal, GQA). Block: 128 q rows x one head; 8 waves each
// own 16 q rows. Double-buffered KV tiles of 64 filled by async global->LDS.
// V is consumed PRE-TRANSPOSED in global memory: Vt[HKV*HD][B*S].
// Softmax in exp2 domain (q pre-scaled by log2(e)/sqrt(HD) in RoPE).
// ---------------------------------------------------------------------------
__global__ __launch_bounds__(256) void flash_attn_kernel(const __bf16* __restrict__ Q,
                                                         const __bf16* __restrict__ Kg,
                                                         const __bf16* __restrict__ Vtg,
                                                         __bf16* __restrict__ Og) {
    constexpr int S = 2048, H = 16, HKV = 4, HD = 128;
    constexpr int M = 2 * S;          // B*S rows (V^T column count)
    constexpr int KSTR = 136;         // K tile row stride ([key][dim])
    constexpr int VSTR = 72;          // V^T tile row stride ([dim][key])
    constexpr int PSTR = 72;          // P staging row stride
    constexpr int KELEMS = 64 * KSTR;     // 8704
    constexpr int VELEMS = 128 * VSTR;    // 9216
    // 2x K + 2x V^T + P
    __shared__ __align__(16) __bf16 smem[2 * KELEMS + 2 * VELEMS + 8 * 16 * PSTR];

    const int tid  = threadIdx.x;
    const int lane = tid & 31;
    const int wave = tid >> 5;
    const int half = lane >> 4;
    const int mr   = lane & 15;
    const int nc   = lane & 15;

    const int qTile = blockIdx.x;
    const int h     = blockIdx.y;
    const int b     = blockIdx.z;
    const int g     = h >> 2;              // kv head
    const int qBase = qTile * 128;

    __bf16* Ps = smem + 2 * KELEMS + 2 * VELEMS;

    // --- stage Q tile (128x128) async into smem (reuses the two K buffers) ---
#pragma unroll
    for (int i = 0; i < 8; ++i) {
        int e   = (tid + i * 256) * 8;
        int row = e >> 7;
        int col = e & 127;
        async_copy_b128(&smem[row * KSTR + col],
                        &Q[(size_t)(b * S + qBase + row) * (H * HD) + h * HD + col]);
    }
    wait_async_0();
    __syncthreads();
    Frag16 qf[4];
#pragma unroll
    for (int c = 0; c < 4; ++c) {
        const __bf16* rp = &smem[(wave * 16 + mr) * KSTR + c * 32];
        qf[c].f4[0] = *(const float4*)&rp[8 * half];
        qf[c].f4[1] = *(const float4*)&rp[16 + 8 * half];
    }
    __syncthreads();

    auto issue_kv = [&](int kvBase, int buf) {
        __bf16* Ksb = smem + buf * KELEMS;
        __bf16* Vtb = smem + 2 * KELEMS + buf * VELEMS;
#pragma unroll
        for (int i = 0; i < 4; ++i) {
            int e = (tid + i * 256) * 8;
            {   // K: [key][dim] direct copy
                int key = e >> 7, dim = e & 127;
                async_copy_b128(&Ksb[key * KSTR + dim],
                                &Kg[(size_t)(b * S + kvBase + key) * (HKV * HD) + g * HD + dim]);
            }
            {   // V^T: [dim][key] direct copy from pre-transposed global
                int dim = e >> 6, key = e & 63;
                async_copy_b128(&Vtb[dim * VSTR + key],
                                &Vtg[(size_t)(g * HD + dim) * M + b * S + kvBase + key]);
            }
        }
    };

    v8f Oacc[8] = {};
    float mrun[8], lrun[8];
#pragma unroll
    for (int r = 0; r < 8; ++r) { mrun[r] = -1e30f; lrun[r] = 0.0f; }

    const int nTiles = (qBase + 128) / 64;
    issue_kv(0, 0);
    for (int t = 0; t < nTiles; ++t) {
        const int buf    = t & 1;
        const int kvBase = t * 64;
        if (t + 1 < nTiles) { issue_kv((t + 1) * 64, buf ^ 1); wait_async_8(); }
        else                { wait_async_0(); }
        __syncthreads();
        const __bf16* Ks = smem + buf * KELEMS;
        const __bf16* Vt = smem + 2 * KELEMS + buf * VELEMS;

        // --- scores S = Q @ K^T (16 x 64), causal masked (exp2 domain) ---
        v8f sf[4];
#pragma unroll
        for (int ns = 0; ns < 4; ++ns) {
            v8f s = {};
#pragma unroll
            for (int c = 0; c < 4; ++c) {
                Frag16 kf;
                const __bf16* rp = &Ks[(ns * 16 + nc) * KSTR + c * 32];
                kf.f4[0] = *(const float4*)&rp[8 * half];
                kf.f4[1] = *(const float4*)&rp[16 + 8 * half];
                s = __builtin_amdgcn_wmma_f32_16x16x32_bf16(
                    false, qf[c].v, false, kf.v, (short)0, s, false, false);
            }
            int kcol = kvBase + ns * 16 + nc;
#pragma unroll
            for (int r = 0; r < 8; ++r) {
                int qrow = qBase + wave * 16 + r + 8 * half;
                if (kcol > qrow) s[r] = -1e30f;
            }
            sf[ns] = s;
        }

        // --- online softmax (base-2) ---
        float mnew[8], alpha[8], psum[8];
#pragma unroll
        for (int r = 0; r < 8; ++r) {
            float mx = fmaxf(fmaxf(sf[0][r], sf[1][r]), fmaxf(sf[2][r], sf[3][r]));
#pragma unroll
            for (int off = 1; off < 16; off <<= 1)
                mx = fmaxf(mx, __shfl_xor(mx, off, 32));
            mnew[r]  = fmaxf(mrun[r], mx);
            alpha[r] = exp2f(mrun[r] - mnew[r]);
            mrun[r]  = mnew[r];
            psum[r]  = 0.0f;
        }
#pragma unroll
        for (int ns = 0; ns < 4; ++ns)
#pragma unroll
            for (int r = 0; r < 8; ++r) {
                float p = exp2f(sf[ns][r] - mnew[r]);
                psum[r] += p;
                Ps[(wave * 16 + r + 8 * half) * PSTR + ns * 16 + nc] = (__bf16)p;
            }
#pragma unroll
        for (int r = 0; r < 8; ++r) {
            float ps = psum[r];
#pragma unroll
            for (int off = 1; off < 16; off <<= 1)
                ps += __shfl_xor(ps, off, 32);
            lrun[r] = lrun[r] * alpha[r] + ps;
        }
#pragma unroll
        for (int ns = 0; ns < 8; ++ns)
#pragma unroll
            for (int r = 0; r < 8; ++r)
                Oacc[ns][r] *= alpha[r];

        // --- O += P @ V ---
        Frag16 pf[2];
#pragma unroll
        for (int c = 0; c < 2; ++c) {
            const __bf16* rp = &Ps[(wave * 16 + mr) * PSTR + c * 32];
            pf[c].f4[0] = *(const float4*)&rp[8 * half];
            pf[c].f4[1] = *(const float4*)&rp[16 + 8 * half];
        }
#pragma unroll
        for (int ns = 0; ns < 8; ++ns)
#pragma unroll
            for (int c = 0; c < 2; ++c) {
                Frag16 vf;
                const __bf16* rp = &Vt[(ns * 16 + nc) * VSTR + c * 32];
                vf.f4[0] = *(const float4*)&rp[8 * half];
                vf.f4[1] = *(const float4*)&rp[16 + 8 * half];
                Oacc[ns] = __builtin_amdgcn_wmma_f32_16x16x32_bf16(
                    false, pf[c].v, false, vf.v, (short)0, Oacc[ns], false, false);
            }
        __syncthreads();
    }

    // --- finalize: O /= l, write bf16 [B*S, H*HD] ---
#pragma unroll
    for (int ns = 0; ns < 8; ++ns)
#pragma unroll
        for (int r = 0; r < 8; ++r) {
            int qrow = qBase + wave * 16 + r + 8 * half;
            float o  = Oacc[ns][r] / lrun[r];
            Og[(size_t)(b * S + qrow) * (H * HD) + h * HD + ns * 16 + nc] = (__bf16)o;
        }
}

// ---------------------------------------------------------------------------
extern "C" void kernel_launch(void* const* d_in, const int* in_sizes, int n_in,
                              void* d_out, int out_size, void* d_ws, size_t ws_size,
                              hipStream_t stream) {
    (void)in_sizes; (void)n_in; (void)out_size; (void)ws_size;
    const int B = 2, S = 2048, DIM = 2048, H = 16, HKV = 4, HD = 128;
    const int M = B * S;                         // 4096
    const int NKV = HKV * HD;                    // 512

    const float* x  = (const float*)d_in[0];
    const float* cs = (const float*)d_in[1];
    const float* sn = (const float*)d_in[2];
    const float* wq = (const float*)d_in[3];
    const float* wk = (const float*)d_in[4];
    const float* wv = (const float*)d_in[5];
    const float* wo = (const float*)d_in[6];

    char* p = (char*)d_ws;
    auto alloc = [&](size_t elems) {
        __bf16* r = (__bf16*)p;
        p += elems * sizeof(__bf16);
        return r;
    };
    __bf16* xb  = alloc((size_t)M * DIM);
    __bf16* wqT = alloc((size_t)DIM * DIM);      // [N][K]
    __bf16* wkT = alloc((size_t)NKV * DIM);
    __bf16* wvT = alloc((size_t)NKV * DIM);
    __bf16* woT = alloc((size_t)DIM * DIM);
    __bf16* qb  = alloc((size_t)M * DIM);
    __bf16* kb  = alloc((size_t)M * NKV);
    __bf16* vtb = alloc((size_t)NKV * M);        // V^T [NKV][M]
    __bf16* ab  = alloc((size_t)M * DIM);

    {   // x: plain convert
        size_t n = (size_t)M * DIM;
        f32_to_bf16_kernel<<<dim3((unsigned)((n + 255) / 256)), 256, 0, stream>>>(x, xb, (int)n);
    }
    // weights: transpose + convert to [N][K]
    transpose_cvt_kernel<<<dim3(DIM / 32, DIM / 32), 256, 0, stream>>>(wq, wqT, DIM, DIM);
    transpose_cvt_kernel<<<dim3(NKV / 32, DIM / 32), 256, 0, stream>>>(wk, wkT, DIM, NKV);
    transpose_cvt_kernel<<<dim3(NKV / 32, DIM / 32), 256, 0, stream>>>(wv, wvT, DIM, NKV);
    transpose_cvt_kernel<<<dim3(DIM / 32, DIM / 32), 256, 0, stream>>>(wo, woT, DIM, DIM);

    // projections (q, k normal layout; v written transposed [NKV][M])
    gemm_bf16_kernel<0><<<dim3(DIM / 128, M / 128), 256, 0, stream>>>(xb, wqT, qb, M, DIM, DIM);
    gemm_bf16_kernel<0><<<dim3(NKV / 128, M / 128), 256, 0, stream>>>(xb, wkT, kb, M, NKV, DIM);
    gemm_bf16_kernel<2><<<dim3(NKV / 128, M / 128), 256, 0, stream>>>(xb, wvT, vtb, M, NKV, DIM);

    // RoPE: q gets log2(e)/sqrt(HD) folded in (exp2-domain flash softmax)
    const float qscale = 0.08838834764831845f * 1.4426950408889634f;
    {
        int total = M * H * (HD / 2);
        rope_kernel<<<(total + 255) / 256, 256, 0, stream>>>(qb, cs, sn, H, total, qscale);
    }
    {
        int total = M * HKV * (HD / 2);
        rope_kernel<<<(total + 255) / 256, 256, 0, stream>>>(kb, cs, sn, HKV, total, 1.0f);
    }

    // causal GQA flash attention
    flash_attn_kernel<<<dim3(S / 128, H, B), 256, 0, stream>>>(qb, kb, vtb, ab);

    // output projection (writes f32 directly to d_out)
    gemm_bf16_kernel<1><<<dim3(DIM / 128, M / 128), 256, 0, stream>>>(ab, woT, (float*)d_out, M, DIM, DIM);
}
